// CrissCrossAttention_48206712930378
// MI455X (gfx1250) — compile-verified, matrix operand order checked by reading, not run on hardware
//
#include <hip/hip_runtime.h>

// ---------------------------------------------------------------------------
// CrissCross attention for MI455X (gfx1250).
// bf16 WMMA pipeline. All GEMM operands are laid out K-contiguous so every
// 16x32 fragment is exactly two 16-byte global loads (global_load_b128),
// feeding v_wmma_f32_16x16x32_bf16 with register blocking (TMxTN tiles).
// ---------------------------------------------------------------------------

typedef __attribute__((ext_vector_type(16))) __bf16 v16bf;
typedef __attribute__((ext_vector_type(8)))  float  v8f;
typedef unsigned short ushort_t;
typedef unsigned int   uint_t;
typedef __attribute__((ext_vector_type(4)))  uint_t v4u;

__device__ __forceinline__ ushort_t f32_to_bf16(float f) {
    union { float f; unsigned int u; } c; c.f = f;
    unsigned int u = c.u;
    u += 0x7FFFu + ((u >> 16) & 1u);   // round-to-nearest-even
    return (ushort_t)(u >> 16);
}
__device__ __forceinline__ uint_t pack_bf16x2(float lo, float hi) {
    return (uint_t)f32_to_bf16(lo) | ((uint_t)f32_to_bf16(hi) << 16);
}

union FragBF { v16bf v; v4u q[2]; ushort_t u[16]; };

// ---------------------------------------------------------------------------
// Blocked tile GEMM, K-contiguous operands (element stride 1 along K).
// A[m,k] = A[m*sAm + k], B[k,n] = B[n*sBn + k]; K % 32 == 0; 16B alignment.
// Fragment (CDNA5 ISA): lane row m=lane&15 (hi=lane>>4);
//   elem e -> K = hi*8 + e + (e>=8?8:0)  == two contiguous 8-elem runs.
// ---------------------------------------------------------------------------
template<int TM, int TN, int K, long sAm, long sBn>
__device__ __forceinline__ void tile_gemm_v(const ushort_t* __restrict__ A,
                                            const ushort_t* __restrict__ B,
                                            v8f (&acc)[TM][TN])
{
    const int lane = (int)(threadIdx.x & 31u);
    const int hi   = lane >> 4;
    const int mn   = lane & 15;

    for (int k0 = 0; k0 < K; k0 += 32) {
        if (k0 + 32 < K) {  // uniform prefetch hint for next K-step
            __builtin_prefetch(A + (long)mn * sAm + (k0 + 32), 0, 0);
            __builtin_prefetch(B + (long)mn * sBn + (k0 + 32), 0, 0);
        }
        FragBF a[TM], b[TN];
#pragma unroll
        for (int tm = 0; tm < TM; ++tm) {
            const ushort_t* pa = A + (long)(tm * 16 + mn) * sAm + k0 + hi * 8;
            a[tm].q[0] = *(const v4u*)pa;
            a[tm].q[1] = *(const v4u*)(pa + 16);
        }
#pragma unroll
        for (int tn = 0; tn < TN; ++tn) {
            const ushort_t* pb = B + (long)(tn * 16 + mn) * sBn + k0 + hi * 8;
            b[tn].q[0] = *(const v4u*)pb;
            b[tn].q[1] = *(const v4u*)(pb + 16);
        }
#pragma unroll
        for (int tm = 0; tm < TM; ++tm)
#pragma unroll
            for (int tn = 0; tn < TN; ++tn)
                acc[tm][tn] = __builtin_amdgcn_wmma_f32_16x16x32_bf16(
                    false, a[tm].v, false, b[tn].v,
                    (short)0, acc[tm][tn], false, false);
    }
}

template<int TM, int TN>
__device__ __forceinline__ void zero_acc(v8f (&acc)[TM][TN]) {
#pragma unroll
    for (int tm = 0; tm < TM; ++tm)
#pragma unroll
        for (int tn = 0; tn < TN; ++tn)
#pragma unroll
            for (int r = 0; r < 8; ++r) acc[tm][tn][r] = 0.f;
}

// ---------------------------------------------------------------------------
// Problem constants
// ---------------------------------------------------------------------------
#define BB 16
#define CC 512
#define CR 64
#define HH 96
#define WW 96
#define PP (HH * WW)          // 9216
#define LG 192                // logits per pixel = H + W

// ---------------------------------------------------------------------------
// Stage 0a: weights f32 -> bf16 (tiny)
// ---------------------------------------------------------------------------
__global__ void k_cvt_w(const float* __restrict__ w, ushort_t* __restrict__ wb, int n)
{
    const int i = (int)(blockIdx.x * blockDim.x + threadIdx.x);
    if (i < n) wb[i] = f32_to_bf16(w[i]);
}

// ---------------------------------------------------------------------------
// Stage 0b: x[b,c,p] f32 -> xT[b,p,c] bf16 (LDS-tiled transpose+convert)
// block (32,8), grid = BB * (CC/32) * (PP/32)
// ---------------------------------------------------------------------------
__global__ void k_cvt_xT(const float* __restrict__ x, ushort_t* __restrict__ xT)
{
    int t = (int)blockIdx.x;
    const int pt = t % (PP / 32); t /= (PP / 32);
    const int ct = t % (CC / 32);
    const int b  = t / (CC / 32);

    __shared__ ushort_t tile[32][33];
    const int tx = (int)threadIdx.x, ty = (int)threadIdx.y;

    const float* src = x + (long)b * CC * PP + (long)(ct * 32) * PP + (long)pt * 32;
#pragma unroll
    for (int r = 0; r < 4; ++r) {
        const int c = ty * 4 + r;
        tile[c][tx] = f32_to_bf16(src[(long)c * PP + tx]);
    }
    __syncthreads();
    ushort_t* dst = xT + ((long)b * PP + (long)pt * 32) * CC + ct * 32;
#pragma unroll
    for (int r = 0; r < 4; ++r) {
        const int p = ty * 4 + r;
        dst[(long)p * CC + tx] = tile[tx][p];
    }
}

// ---------------------------------------------------------------------------
// Stage 1a: q/k projection -> channel-last qT[b,p,c] (c in [0,64))
// D[o,p] = sum_c Wb[o,c]*xT[b,p,c] + bias[o].  64x64 per wave.
// grid.x = BB * (PP/64)
// ---------------------------------------------------------------------------
__global__ __launch_bounds__(32) void k_proj_qk(const ushort_t* __restrict__ xT,
                                                const ushort_t* __restrict__ Wb,
                                                const float* __restrict__ bias,
                                                ushort_t* __restrict__ qT)
{
    const int pt = (int)blockIdx.x % (PP / 64);
    const int b  = (int)blockIdx.x / (PP / 64);

    const ushort_t* A = Wb;                                         // [m=o,k=c] sAm=CC
    const ushort_t* B = xT + ((long)b * PP + (long)pt * 64) * CC;   // [n=p,k=c] sBn=CC

    v8f acc[4][4];
    zero_acc(acc);
    tile_gemm_v<4, 4, CC, (long)CC, (long)CC>(A, B, acc);

    const int lane = (int)(threadIdx.x & 31u);
    const int hi = lane >> 4, n = lane & 15;
#pragma unroll
    for (int tn = 0; tn < 4; ++tn) {
        const int p = pt * 64 + tn * 16 + n;
        ushort_t* dst = qT + ((long)b * PP + p) * CR;
#pragma unroll
        for (int tm = 0; tm < 4; ++tm) {
            const int o0 = tm * 16 + hi * 8;
            v4u pk;
#pragma unroll
            for (int h2 = 0; h2 < 4; ++h2)
                pk[h2] = pack_bf16x2(acc[tm][tn][2 * h2]     + bias[o0 + 2 * h2],
                                     acc[tm][tn][2 * h2 + 1] + bias[o0 + 2 * h2 + 1]);
            *(v4u*)(dst + o0) = pk;   // 16B contiguous store
        }
    }
}

// ---------------------------------------------------------------------------
// Stage 1b: v projection (transposed orientation) -> natural vb[b,o,p]
// D[p,o] = sum_c xT[b,p,c]*Wvb[o,c] + bias[o].  64x64 per wave.
// grid.x = BB * (PP/64) * (CC/64)
// ---------------------------------------------------------------------------
__global__ __launch_bounds__(32) void k_proj_v(const ushort_t* __restrict__ xT,
                                               const ushort_t* __restrict__ Wvb,
                                               const float* __restrict__ bias,
                                               ushort_t* __restrict__ vb)
{
    int t = (int)blockIdx.x;
    const int ob = t % (CC / 64); t /= (CC / 64);
    const int pb = t % (PP / 64);
    const int b  = t / (PP / 64);

    const ushort_t* A = xT + ((long)b * PP + (long)pb * 64) * CC;   // [m=p,k=c] sAm=CC
    const ushort_t* B = Wvb + (long)ob * 64 * CC;                   // [n=o,k=c] sBn=CC

    v8f acc[4][4];
    zero_acc(acc);
    tile_gemm_v<4, 4, CC, (long)CC, (long)CC>(A, B, acc);

    const int lane = (int)(threadIdx.x & 31u);
    const int hi = lane >> 4, n = lane & 15;
#pragma unroll
    for (int tn = 0; tn < 4; ++tn) {
        const int o = ob * 64 + tn * 16 + n;
        const float bo = bias[o];
        ushort_t* dst = vb + ((long)b * CC + o) * PP + (long)pb * 64;
#pragma unroll
        for (int tm = 0; tm < 4; ++tm) {
            const int p0 = tm * 16 + hi * 8;
            v4u pk;
#pragma unroll
            for (int h2 = 0; h2 < 4; ++h2)
                pk[h2] = pack_bf16x2(acc[tm][tn][2 * h2] + bo,
                                     acc[tm][tn][2 * h2 + 1] + bo);
            *(v4u*)(dst + p0) = pk;   // 16B contiguous store
        }
    }
}

// ---------------------------------------------------------------------------
// Stage 2a: eh[b,i,w,j] = sum_c qT[b,(i,w),c]*kT[b,(j,w),c] -> logits[pix][j]
// 32x32 per wave. grid.x = BB * WW * 9
// ---------------------------------------------------------------------------
__global__ __launch_bounds__(32) void k_eh(const ushort_t* __restrict__ qT,
                                           const ushort_t* __restrict__ kT,
                                           float* __restrict__ logits)
{
    int t = (int)blockIdx.x;
    const int nb = t % 3; t /= 3;
    const int mb = t % 3; t /= 3;
    const int w  = t % WW;
    const int b  = t / WW;

    const ushort_t* A = qT + (long)b * PP * CR + (long)w * CR + (long)mb * 32 * (WW * CR);
    const ushort_t* B = kT + (long)b * PP * CR + (long)w * CR + (long)nb * 32 * (WW * CR);

    v8f acc[2][2];
    zero_acc(acc);
    tile_gemm_v<2, 2, CR, (long)(WW * CR), (long)(WW * CR)>(A, B, acc);

    const int lane = (int)(threadIdx.x & 31u);
    const int hi = lane >> 4, n = lane & 15;
#pragma unroll
    for (int tm = 0; tm < 2; ++tm)
#pragma unroll
        for (int tn = 0; tn < 2; ++tn)
#pragma unroll
            for (int r = 0; r < 8; ++r) {
                const int i = mb * 32 + tm * 16 + hi * 8 + r;
                const int j = nb * 32 + tn * 16 + n;
                logits[(((long)b * HH + i) * WW + w) * LG + j] = acc[tm][tn][r];
            }
}

// ---------------------------------------------------------------------------
// Stage 2b: ew[b,h,i,j] = sum_c qT[b,(h,i),c]*kT[b,(h,j),c] -> logits[pix][96+j]
// ---------------------------------------------------------------------------
__global__ __launch_bounds__(32) void k_ew(const ushort_t* __restrict__ qT,
                                           const ushort_t* __restrict__ kT,
                                           float* __restrict__ logits)
{
    int t = (int)blockIdx.x;
    const int nb = t % 3; t /= 3;
    const int mb = t % 3; t /= 3;
    const int h  = t % HH;
    const int b  = t / HH;

    const ushort_t* A = qT + (long)b * PP * CR + (long)h * WW * CR + (long)mb * 32 * CR;
    const ushort_t* B = kT + (long)b * PP * CR + (long)h * WW * CR + (long)nb * 32 * CR;

    v8f acc[2][2];
    zero_acc(acc);
    tile_gemm_v<2, 2, CR, (long)CR, (long)CR>(A, B, acc);

    const int lane = (int)(threadIdx.x & 31u);
    const int hi = lane >> 4, n = lane & 15;
#pragma unroll
    for (int tm = 0; tm < 2; ++tm)
#pragma unroll
        for (int tn = 0; tn < 2; ++tn)
#pragma unroll
            for (int r = 0; r < 8; ++r) {
                const int i = mb * 32 + tm * 16 + hi * 8 + r;
                const int j = nb * 32 + tn * 16 + n;
                logits[(((long)b * HH + h) * WW + i) * LG + HH + j] = acc[tm][tn][r];
            }
}

// ---------------------------------------------------------------------------
// Stage 3: softmax over 192 logits per pixel -> bf16. One wave per pixel.
// ---------------------------------------------------------------------------
__global__ void k_softmax(const float* __restrict__ logits,
                          ushort_t* __restrict__ attn)
{
    const int pix  = (int)blockIdx.x * 8 + (int)(threadIdx.x >> 5);
    const int lane = (int)(threadIdx.x & 31u);

    const float* row = logits + (long)pix * LG;
    ushort_t*    out = attn + (long)pix * LG;

    float v[6];
    float mx = -3.4e38f;
#pragma unroll
    for (int tt = 0; tt < 6; ++tt) { v[tt] = row[tt * 32 + lane]; mx = fmaxf(mx, v[tt]); }
#pragma unroll
    for (int m = 16; m >= 1; m >>= 1) mx = fmaxf(mx, __shfl_xor(mx, m, 32));
    float s = 0.f;
#pragma unroll
    for (int tt = 0; tt < 6; ++tt) { v[tt] = __expf(v[tt] - mx); s += v[tt]; }
#pragma unroll
    for (int m = 16; m >= 1; m >>= 1) s += __shfl_xor(s, m, 32);
    const float inv = 1.f / s;
#pragma unroll
    for (int tt = 0; tt < 6; ++tt) out[tt * 32 + lane] = f32_to_bf16(v[tt] * inv);
}

// ---------------------------------------------------------------------------
// Stage 3b: vb[b,c,h,w] -> vhT[b,c,w,h] (per-plane spatial transpose, bf16)
// block (32,8), grid = BB*CC*9
// ---------------------------------------------------------------------------
__global__ void k_trans_v(const ushort_t* __restrict__ vb, ushort_t* __restrict__ vhT)
{
    int t = (int)blockIdx.x;
    const int tw = t % 3; t /= 3;
    const int th = t % 3;
    const int pl = t / 3;   // plane = b*CC + c

    __shared__ ushort_t tile[32][33];
    const int tx = (int)threadIdx.x, ty = (int)threadIdx.y;

    const ushort_t* src = vb + (long)pl * PP + (long)(th * 32) * WW + tw * 32;
#pragma unroll
    for (int r = 0; r < 4; ++r) {
        const int hl = ty * 4 + r;
        tile[hl][tx] = src[(long)hl * WW + tx];
    }
    __syncthreads();
    ushort_t* dst = vhT + (long)pl * PP + (long)(tw * 32) * HH + th * 32;
#pragma unroll
    for (int r = 0; r < 4; ++r) {
        const int wl = ty * 4 + r;
        dst[(long)wl * HH + tx] = tile[tx][wl];
    }
}

// ---------------------------------------------------------------------------
// Stage 4a: out[b,c,i,w] = x[b,c,i,w] + g * sum_j vhT[b,c,w,j]*ah[b,i,w,j]
// 64x32 per wave. grid.x = BB * WW * 8 * 3
// ---------------------------------------------------------------------------
__global__ __launch_bounds__(32) void k_outh(const ushort_t* __restrict__ vhT,
                                             const ushort_t* __restrict__ attn,
                                             const float* __restrict__ x,
                                             const float* __restrict__ gamma,
                                             float* __restrict__ out)
{
    int t = (int)blockIdx.x;
    const int nb = t % 3; t /= 3;
    const int mb = t % 8; t /= 8;
    const int w  = t % WW;
    const int b  = t / WW;
    const float g = gamma[0];

    // A[m=c,k=j] = vhT[b,c,w,j] : sAm=PP, k stride 1
    const ushort_t* A = vhT + (long)b * CC * PP + (long)w * HH + (long)mb * 64 * PP;
    // B[k=j,n=i] = attn[((b*H+i)*W+w)*LG + j] : sBn=W*LG, k stride 1
    const ushort_t* B = attn + (long)b * HH * WW * LG + (long)w * LG + (long)nb * 32 * (WW * LG);

    v8f acc[4][2];
    zero_acc(acc);
    tile_gemm_v<4, 2, HH, (long)PP, (long)(WW * LG)>(A, B, acc);

    const int lane = (int)(threadIdx.x & 31u);
    const int hi = lane >> 4, n = lane & 15;
#pragma unroll
    for (int tm = 0; tm < 4; ++tm)
#pragma unroll
        for (int tn = 0; tn < 2; ++tn)
#pragma unroll
            for (int r = 0; r < 8; ++r) {
                const int c = mb * 64 + tm * 16 + hi * 8 + r;
                const int i = nb * 32 + tn * 16 + n;
                const long oi = (((long)b * CC + c) * HH + i) * WW + w;
                out[oi] = x[oi] + g * acc[tm][tn][r];
            }
}

// ---------------------------------------------------------------------------
// Stage 4b: out[b,c,h,i] += g * sum_j vb[b,c,h,j]*aw[b,h,i,j]
// 64x32 per wave. grid.x = BB * HH * 8 * 3  (after k_outh; disjoint RMW)
// ---------------------------------------------------------------------------
__global__ __launch_bounds__(32) void k_outw(const ushort_t* __restrict__ vb,
                                             const ushort_t* __restrict__ attn,
                                             const float* __restrict__ gamma,
                                             float* __restrict__ out)
{
    int t = (int)blockIdx.x;
    const int nb = t % 3; t /= 3;
    const int mb = t % 8; t /= 8;
    const int h  = t % HH;
    const int b  = t / HH;
    const float g = gamma[0];

    // A[m=c,k=j] = vb[b,c,h,j] : sAm=PP, k stride 1
    const ushort_t* A = vb + (long)b * CC * PP + (long)h * WW + (long)mb * 64 * PP;
    // B[k=j,n=i] = attn[((b*H+h)*W+i)*LG + 96 + j] : sBn=LG, k stride 1
    const ushort_t* B = attn + ((long)b * HH + h) * WW * LG + HH + (long)nb * 32 * LG;

    v8f acc[4][2];
    zero_acc(acc);
    tile_gemm_v<4, 2, WW, (long)PP, (long)LG>(A, B, acc);

    const int lane = (int)(threadIdx.x & 31u);
    const int hi = lane >> 4, n = lane & 15;
#pragma unroll
    for (int tm = 0; tm < 4; ++tm)
#pragma unroll
        for (int tn = 0; tn < 2; ++tn)
#pragma unroll
            for (int r = 0; r < 8; ++r) {
                const int c = mb * 64 + tm * 16 + hi * 8 + r;
                const int i = nb * 32 + tn * 16 + n;
                const long oi = (((long)b * CC + c) * HH + h) * WW + i;
                out[oi] += g * acc[tm][tn][r];
            }
}

// ---------------------------------------------------------------------------
// Host launcher
// ---------------------------------------------------------------------------
extern "C" void kernel_launch(void* const* d_in, const int* in_sizes, int n_in,
                              void* d_out, int out_size, void* d_ws, size_t ws_size,
                              hipStream_t stream)
{
    (void)in_sizes; (void)n_in; (void)out_size; (void)ws_size;

    const float* x     = (const float*)d_in[0];
    const float* Wq    = (const float*)d_in[1];
    const float* bq    = (const float*)d_in[2];
    const float* Wk    = (const float*)d_in[3];
    const float* bk    = (const float*)d_in[4];
    const float* Wv    = (const float*)d_in[5];
    const float* bv    = (const float*)d_in[6];
    const float* gamma = (const float*)d_in[7];
    float* out = (float*)d_out;

    // Workspace layout (bf16 as ushort). Region 0 (151MB) is time-shared:
    //   stage 0-1: xT ; stage 2-3: f32 logits (113MB) ; stage 3b-4: vhT (151MB)
    const long X_ELEMS  = (long)BB * CC * PP;   // 75,497,472
    const long QK_ELEMS = (long)BB * CR * PP;   //  9,437,184
    const long PIX      = (long)BB * HH * WW;   //    147,456

    ushort_t* xT    = (ushort_t*)d_ws;
    float*    logit = (float*)d_ws;
    ushort_t* vhT   = (ushort_t*)d_ws;
    ushort_t* qT    = xT + X_ELEMS;
    ushort_t* kT    = qT + QK_ELEMS;
    ushort_t* vb    = kT + QK_ELEMS;
    ushort_t* attn  = vb + X_ELEMS;
    ushort_t* Wqb   = attn + PIX * LG;
    ushort_t* Wkb   = Wqb + CR * CC;
    ushort_t* Wvb   = Wkb + CR * CC;
    // peak ws: 151 + 18.9 + 18.9 + 151 + 56.6 + 0.6 ~= 397 MB

    const dim3 wv(32);
    const dim3 t32x8(32, 8);

    // Stage 0: weight + input conversion / re-layout
    k_cvt_w<<<dim3((CR * CC) / 256), dim3(256), 0, stream>>>(Wq, Wqb, CR * CC);
    k_cvt_w<<<dim3((CR * CC) / 256), dim3(256), 0, stream>>>(Wk, Wkb, CR * CC);
    k_cvt_w<<<dim3((CC * CC) / 256), dim3(256), 0, stream>>>(Wv, Wvb, CC * CC);
    k_cvt_xT<<<dim3(BB * (CC / 32) * (PP / 32)), t32x8, 0, stream>>>(x, xT);

    // Stage 1: projections
    k_proj_qk<<<dim3(BB * (PP / 64)), wv, 0, stream>>>(xT, Wqb, bq, qT);
    k_proj_qk<<<dim3(BB * (PP / 64)), wv, 0, stream>>>(xT, Wkb, bk, kT);
    k_proj_v<<<dim3(BB * (PP / 64) * (CC / 64)), wv, 0, stream>>>(xT, Wvb, bv, vb);

    // Stage 2: attention logits (region 0 reused; xT dead)
    k_eh<<<dim3(BB * WW * 9), wv, 0, stream>>>(qT, kT, logit);
    k_ew<<<dim3(BB * HH * 9), wv, 0, stream>>>(qT, kT, logit);

    // Stage 3: softmax, then spatial transpose of v (region 0 reused; logits dead)
    k_softmax<<<dim3((unsigned)(PIX / 8)), dim3(256), 0, stream>>>(logit, attn);
    k_trans_v<<<dim3(BB * CC * 9), t32x8, 0, stream>>>(vb, vhT);

    // Stage 4: aggregation + residual epilogue
    k_outh<<<dim3(BB * WW * 24), wv, 0, stream>>>(vhT, attn, x, gamma, out);
    k_outw<<<dim3(BB * HH * 24), wv, 0, stream>>>(vb, attn, gamma, out);
}
